// ShiftWindowMSA_88596585382329
// MI455X (gfx1250) — matrix operand order repaired and checked by reading, not compiled, not run on hardware
//
#include <hip/hip_runtime.h>
#include <hip/hip_bf16.h>

#define WSZ   7
#define SHF   3
#define NHEAD 6
#define HDIM  32
#define CCH   192
#define HH    56
#define BB    32
#define NWIN  2048          // 32 * 8 * 8
#define NTOK  49
#define NPAD  64
#define MROWS 100352        // 2048*49  (divisible by 16: 6272 tiles)
#define MPAD  131072        // 2048*64

typedef __attribute__((ext_vector_type(16))) _Float16 v16h;
typedef __attribute__((ext_vector_type(8)))  _Float16 v8h;
typedef __attribute__((ext_vector_type(8)))  float    v8f;

__device__ __forceinline__ v8f wmma16(v16h a, v16h b, v8f c) {
  return __builtin_amdgcn_wmma_f32_16x16x32_f16(false, a, false, b, (short)0, c,
                                                false, false);
}

// A-fragment (16x32 f16): lane<16 -> K = kk+{0..7, 16..23}; lane>=16 -> +8
__device__ __forceinline__ v16h load_a_f16(const _Float16* __restrict__ row,
                                           int kk, int lane) {
  const int b = (lane & 16) ? 8 : 0;
  v8h lo = *(const v8h*)(row + kk + b);
  v8h hi = *(const v8h*)(row + kk + b + 16);
  v16h r;
#pragma unroll
  for (int i = 0; i < 8; ++i) { r[i] = lo[i]; r[i + 8] = hi[i]; }
  return r;
}

__device__ __forceinline__ v16h load_a_lds(const _Float16* row, int kk, int lane) {
  const int b = (lane & 16) ? 8 : 0;
  v8h lo = *(const v8h*)(row + kk + b);
  v8h hi = *(const v8h*)(row + kk + b + 16);
  v16h r;
#pragma unroll
  for (int i = 0; i < 8; ++i) { r[i] = lo[i]; r[i + 8] = hi[i]; }
  return r;
}

// A-fragment gathered from fp32 with convert
__device__ __forceinline__ v16h load_a_f32(const float* __restrict__ row,
                                           int kk, int lane) {
  const int b = (lane & 16) ? 8 : 0;
  const float* p0 = row + kk + b;
  const float* p1 = p0 + 16;
  v16h r;
#pragma unroll
  for (int i = 0; i < 8; ++i) {
    r[i]     = (_Float16)p0[i];
    r[i + 8] = (_Float16)p1[i];
  }
  return r;
}

// B-fragment (32x16 f16), nrow points at W[n][0] (K-contiguous)
__device__ __forceinline__ v16h load_b_f16(const _Float16* __restrict__ nrow,
                                           int kk, int lane) {
  const int b = (lane & 16) ? 8 : 0;
  v8h lo = *(const v8h*)(nrow + kk + b);
  v8h hi = *(const v8h*)(nrow + kk + b + 16);
  v16h r;
#pragma unroll
  for (int i = 0; i < 8; ++i) { r[i] = lo[i]; r[i + 8] = hi[i]; }
  return r;
}

__device__ __forceinline__ int region_label(int r) {
  return (r < 49) ? 0 : ((r < 53) ? 1 : 2);
}

// ---------------- kernel 0: weight fp32 -> f16 -----------------------------
__global__ __launch_bounds__(256) void cvt_weights(
    const float* __restrict__ qkvw, const float* __restrict__ projw,
    _Float16* __restrict__ wq, _Float16* __restrict__ wp) {
  int i = blockIdx.x * 256 + threadIdx.x;
  if (i < 3 * CCH * CCH) wq[i] = (_Float16)qkvw[i];
  if (i < CCH * CCH)     wp[i] = (_Float16)projw[i];
}

// ---------------- kernel 1: fused roll+window-partition + QKV GEMM ---------
// M' = 131072 (64 rows/window), N = 576, K = 192. One wave = 16x64 strip.
__global__ __launch_bounds__(256) void qkv_gemm(
    const float* __restrict__ x, const _Float16* __restrict__ wq,
    const float* __restrict__ qkvb,
    _Float16* __restrict__ qws, _Float16* __restrict__ kws,
    _Float16* __restrict__ vws) {
  const int lane = threadIdx.x & 31;
  const int wave = threadIdx.x >> 5;
  const int task = blockIdx.x * 8 + wave;       // 73728 tasks
  const int mt = task / 9;
  const int nb = task % 9;
  const int m0 = mt * 16;
  const int n0 = nb * 64;

  // gather source row for this lane's A row (shifted image, window order)
  {
  }
  const int mg    = m0 + (lane & 15);
  const int win   = mg >> 6;
  const int local = mg & 63;
  const int p     = (local < NTOK) ? local : 0;
  const int bimg  = win >> 6;
  const int widx  = win & 63;
  const int wy = widx >> 3, wx = widx & 7;
  int r = wy * WSZ + p / WSZ + SHF; if (r >= HH) r -= HH;
  int c = wx * WSZ + p % WSZ + SHF; if (c >= HH) c -= HH;
  const float* arow = x + (size_t)(bimg * (HH * HH) + r * HH + c) * CCH;

  v8f acc[4] = {v8f{}, v8f{}, v8f{}, v8f{}};
#pragma unroll
  for (int kk = 0; kk < CCH; kk += 32) {
    v16h a = load_a_f32(arow, kk, lane);
#pragma unroll
    for (int t = 0; t < 4; ++t) {
      const int n = n0 + t * 16 + (lane & 15);
      v16h b = load_b_f16(wq + (size_t)n * CCH, kk, lane);
      acc[t] = wmma16(a, b, acc[t]);
    }
  }

  const float scale = 0.17677669529663687f;  // 32^-0.5
#pragma unroll
  for (int t = 0; t < 4; ++t) {
    const int n    = n0 + t * 16 + (lane & 15);
    const int cls  = n / CCH;          // 0=q 1=k 2=v
    const int rem  = n % CCH;
    const int head = rem >> 5;
    const int d    = rem & 31;
    const float bias = qkvb[n];
#pragma unroll
    for (int rr = 0; rr < 8; ++rr) {
      const int mloc = m0 + rr + ((lane & 16) ? 8 : 0);
      const int w2   = mloc >> 6;
      const int loc  = mloc & 63;
      float v = acc[t][rr] + bias;
      if (cls == 0) v *= scale;
      if (loc >= NTOK) v = 0.0f;       // zero the padded rows
      const _Float16 hv = (_Float16)v;
      const size_t wh = (size_t)w2 * NHEAD + head;
      if (cls == 0)      qws[(wh * NPAD + loc) * HDIM + d] = hv;
      else if (cls == 1) kws[(wh * NPAD + loc) * HDIM + d] = hv;
      else               vws[(wh * HDIM + d) * NPAD + loc] = hv;  // transposed
    }
  }
}

// ---------------- kernel 2: fused windowed attention -----------------------
// One block per (window, head): 4 waves x 16 query rows.
__global__ __launch_bounds__(128) void attn_kernel(
    const _Float16* __restrict__ qws, const _Float16* __restrict__ kws,
    const _Float16* __restrict__ vws, const float* __restrict__ rpb,
    _Float16* __restrict__ ows) {
  __shared__ float    srow[4][16][NPAD];     // 16 KB
  __shared__ _Float16 prow[4][16][NPAD];     //  8 KB
  __shared__ float    rpb_s[169 * NHEAD];    //  4 KB

  const int wh   = blockIdx.x;               // window*6 + head
  const int win  = wh / NHEAD;
  const int head = wh % NHEAD;
  const int widx = win & 63;
  const int wy = widx >> 3, wx = widx & 7;
  const int lane = threadIdx.x & 31;
  const int wave = threadIdx.x >> 5;
  const int m0   = wave * 16;

  for (int i = threadIdx.x; i < 169 * NHEAD; i += 128) rpb_s[i] = rpb[i];

  const _Float16* qbase = qws + (size_t)wh * (NPAD * HDIM);
  const _Float16* kbase = kws + (size_t)wh * (NPAD * HDIM);
  const _Float16* vbase = vws + (size_t)wh * (HDIM * NPAD);

  // q A-fragment: full head_dim (K=32) in one fragment
  v16h qa = load_a_f16(qbase + (size_t)(m0 + (lane & 15)) * HDIM, 0, lane);
  __syncthreads();   // rpb_s ready

  // S = q k^T + bias + mask -> LDS
#pragma unroll
  for (int t = 0; t < 4; ++t) {
    const int j = t * 16 + (lane & 15);      // key index (column)
    v16h kb = load_b_f16(kbase + (size_t)j * HDIM, 0, lane);
    v8f s = {};
    s = wmma16(qa, kb, s);
    const int jy = j / WSZ, jx = j % WSZ;
    const int lj = (j < NTOK)
        ? region_label(wy * WSZ + jy) * 3 + region_label(wx * WSZ + jx) : 0;
    const int cj = (j < NTOK)
        ? ((48 - j) / WSZ) * 13 + ((48 - j) % WSZ) : 0;
#pragma unroll
    for (int rr = 0; rr < 8; ++rr) {
      const int i  = m0 + rr + ((lane & 16) ? 8 : 0);
      float val = -1e30f;
      if (i < NTOK && j < NTOK) {
        const int iy = i / WSZ, ix = i % WSZ;
        const int ci = iy * 13 + ix;
        val = s[rr] + rpb_s[(ci + cj) * NHEAD + head];
        const int li = region_label(wy * WSZ + iy) * 3 +
                       region_label(wx * WSZ + ix);
        if (li != lj) val -= 100.0f;
      }
      srow[wave][rr + ((lane & 16) ? 8 : 0)][j] = val;
    }
  }
  __syncthreads();

  // softmax: one thread per row (64 rows)
  if (threadIdx.x < 64) {
    const int g = threadIdx.x;
    const int wv = g >> 4, rl = g & 15;
    if (g < NTOK) {
      float mx = -1e30f;
      for (int jc = 0; jc < NPAD; ++jc) mx = fmaxf(mx, srow[wv][rl][jc]);
      float sum = 0.0f;
      float e[NPAD];
      for (int jc = 0; jc < NPAD; ++jc) {
        e[jc] = __expf(srow[wv][rl][jc] - mx);
        sum += e[jc];
      }
      const float inv = 1.0f / (sum + 1e-20f);
      for (int jc = 0; jc < NPAD; ++jc)
        prow[wv][rl][jc] = (_Float16)(e[jc] * inv);
    } else {
      for (int jc = 0; jc < NPAD; ++jc) prow[wv][rl][jc] = (_Float16)0.0f;
    }
  }
  __syncthreads();

  // O = P V  (K = 64 keys = 2 WMMA steps; 2 d-tiles)
  v8f o[2] = {v8f{}, v8f{}};
#pragma unroll
  for (int kt = 0; kt < 2; ++kt) {
    const int kk = kt * 32;
    v16h pa = load_a_lds(&prow[wave][lane & 15][0], kk, lane);
#pragma unroll
    for (int dt = 0; dt < 2; ++dt) {
      const int d = dt * 16 + (lane & 15);
      v16h vb = load_b_f16(vbase + (size_t)d * NPAD, kk, lane);
      o[dt] = wmma16(pa, vb, o[dt]);
    }
  }
#pragma unroll
  for (int dt = 0; dt < 2; ++dt) {
    const int d = dt * 16 + (lane & 15);
#pragma unroll
    for (int rr = 0; rr < 8; ++rr) {
      const int m = m0 + rr + ((lane & 16) ? 8 : 0);
      if (m < NTOK)
        ows[((size_t)win * NTOK + m) * CCH + head * HDIM + d] =
            (_Float16)o[dt][rr];
    }
  }
}

// ---------------- kernel 3: proj GEMM + window-reverse + roll scatter ------
// M = 100352, N = 192, K = 192. One wave = 16x64 strip (3 n-blocks).
__global__ __launch_bounds__(256) void proj_gemm(
    const _Float16* __restrict__ aws, const _Float16* __restrict__ wp,
    const float* __restrict__ pb, float* __restrict__ out) {
  const int lane = threadIdx.x & 31;
  const int wave = threadIdx.x >> 5;
  const int task = blockIdx.x * 8 + wave;      // 18816 tasks
  const int mt = task / 3;
  const int nb = task % 3;
  const int m0 = mt * 16;
  const int n0 = nb * 64;

  const _Float16* arow = aws + (size_t)(m0 + (lane & 15)) * CCH;
  v8f acc[4] = {v8f{}, v8f{}, v8f{}, v8f{}};
#pragma unroll
  for (int kk = 0; kk < CCH; kk += 32) {
    v16h a = load_a_f16(arow, kk, lane);
#pragma unroll
    for (int t = 0; t < 4; ++t) {
      const int n = n0 + t * 16 + (lane & 15);
      v16h b = load_b_f16(wp + (size_t)n * CCH, kk, lane);
      acc[t] = wmma16(a, b, acc[t]);
    }
  }

#pragma unroll
  for (int t = 0; t < 4; ++t) {
    const int n = n0 + t * 16 + (lane & 15);
    const float bias = pb[n];
#pragma unroll
    for (int rr = 0; rr < 8; ++rr) {
      const int m   = m0 + rr + ((lane & 16) ? 8 : 0);
      const int win = m / NTOK;
      const int p   = m % NTOK;
      const int bimg = win >> 6;
      const int widx = win & 63;
      const int wy = widx >> 3, wx = widx & 7;
      int r = wy * WSZ + p / WSZ + SHF; if (r >= HH) r -= HH;   // roll back
      int c = wx * WSZ + p % WSZ + SHF; if (c >= HH) c -= HH;
      out[((size_t)(bimg * (HH * HH) + r * HH + c)) * CCH + n] =
          acc[t][rr] + bias;
    }
  }
}

extern "C" void kernel_launch(void* const* d_in, const int* in_sizes, int n_in,
                              void* d_out, int out_size, void* d_ws,
                              size_t ws_size, hipStream_t stream) {
  const float* x     = (const float*)d_in[0];   // (32, 3136, 192)
  const float* rpb   = (const float*)d_in[1];   // (169, 6)
  const float* qkvw  = (const float*)d_in[2];   // (576, 192)
  const float* qkvb  = (const float*)d_in[3];   // (576,)
  const float* projw = (const float*)d_in[4];   // (192, 192)
  const float* projb = (const float*)d_in[5];   // (192,)
  float* out = (float*)d_out;

  // workspace carve-up (256B aligned)
  char* ws = (char*)d_ws;
  size_t off = 0;
  auto carve = [&](size_t bytes) {
    char* p = ws + off;
    off += (bytes + 255) & ~(size_t)255;
    return p;
  };
  _Float16* wq  = (_Float16*)carve((size_t)3 * CCH * CCH * 2);
  _Float16* wp  = (_Float16*)carve((size_t)CCH * CCH * 2);
  _Float16* qws = (_Float16*)carve((size_t)NWIN * NHEAD * NPAD * HDIM * 2);
  _Float16* kws = (_Float16*)carve((size_t)NWIN * NHEAD * NPAD * HDIM * 2);
  _Float16* vws = (_Float16*)carve((size_t)NWIN * NHEAD * HDIM * NPAD * 2);
  _Float16* ows = (_Float16*)carve((size_t)MROWS * CCH * 2);
  (void)ws_size; (void)in_sizes; (void)n_in; (void)out_size;

  cvt_weights<<<(3 * CCH * CCH + 255) / 256, 256, 0, stream>>>(qkvw, projw,
                                                              wq, wp);
  // 131072/16 = 8192 m-tiles * 9 n-blocks = 73728 waves / 8 = 9216 blocks
  qkv_gemm<<<9216, 256, 0, stream>>>(x, wq, qkvb, qws, kws, vws);
  // one block per (window, head)
  attn_kernel<<<NWIN * NHEAD, 128, 0, stream>>>(qws, kws, vws, rpb, ows);
  // 6272 m-tiles * 3 n-blocks = 18816 waves / 8 = 2352 blocks
  proj_gemm<<<2352, 256, 0, stream>>>(ows, wp, projb, out);
}
